// OpenPanguMoEAttention_9620726743829
// MI455X (gfx1250) — matrix operand order repaired and checked by reading, not compiled, not run on hardware
//
#include <hip/hip_runtime.h>

typedef __attribute__((ext_vector_type(16))) __bf16 bf16x16;
typedef __attribute__((ext_vector_type(8)))  float  f32x8;

// ---------- helpers ----------
__device__ __forceinline__ unsigned short f2bf(float f) {
  unsigned u = __float_as_uint(f);
  unsigned r = u + 0x7FFFu + ((u >> 16) & 1u);   // round-to-nearest-even
  return (unsigned short)(r >> 16);
}

union FragBF { unsigned int u32[8]; unsigned short u16[16]; bf16x16 v; };

__device__ __forceinline__ f32x8 wmma_bf16(const FragBF& a, const FragBF& b, f32x8 c) {
  return __builtin_amdgcn_wmma_f32_16x16x32_bf16(false, a.v, false, b.v,
                                                 (short)0, c, false, false);
}

// A fragment (16x32, row-major source, lda in elements). ISA 7.12.2 16-bit A layout.
__device__ __forceinline__ FragBF load_frag_a(const unsigned short* A, int lda) {
  int lane = threadIdx.x & 31;
  int m  = lane & 15;
  int kb = (lane < 16) ? 0 : 8;
  const unsigned short* row = A + m * lda;
  FragBF f;
#pragma unroll
  for (int i = 0; i < 8; ++i) {
    int k = (i < 4) ? (kb + 2 * i) : (kb + 16 + 2 * (i - 4));
    f.u32[i] = *(const unsigned int*)(row + k);
  }
  return f;
}

// B fragment (32x16). Source stored COLUMN-major: Bt[n][k], ldb in elements.
__device__ __forceinline__ FragBF load_frag_b(const unsigned short* Bt, int ldb) {
  int lane = threadIdx.x & 31;
  int n  = lane & 15;
  int kg = (lane < 16) ? 0 : 8;
  const unsigned short* col = Bt + n * ldb + kg;
  FragBF f;
#pragma unroll
  for (int i = 0; i < 8; ++i) f.u32[i] = *(const unsigned int*)(col + ((i < 4) ? 2 * i : 16 + 2 * (i - 4)));
  return f;
}

// ---------- elementwise convert (vectorized: 4 floats -> 4 bf16 per thread) ----------
__global__ void k_f32_to_bf16_v4(const float* __restrict__ x,
                                 unsigned short* __restrict__ y, int n4) {
  int i = blockIdx.x * blockDim.x + threadIdx.x;
  if (i < n4) {
    float4 v = ((const float4*)x)[i];
    union { unsigned short s[4]; uint2 u; } o;
    o.s[0] = f2bf(v.x); o.s[1] = f2bf(v.y);
    o.s[2] = f2bf(v.z); o.s[3] = f2bf(v.w);
    ((uint2*)y)[i] = o.u;
  }
}

// ---------- generic bf16 GEMM: C(MxN,f32) = A(MxK) * B(KxN), all row-major ----------
// BM=128, BN=64, BK=32; 8 waves, wave tile 32x32 (2x2 WMMA tiles, 32 acc VGPRs).
// Double-buffered LDS; prefetch values held in plain uint4 scalars (no lambdas /
// address-taken arrays, so they stay in VGPRs and the load-wait overlaps compute).
#define BM 128
#define BN 64
#define BK 32
#define LDA_S 40
#define LDB_S 40

__global__ __launch_bounds__(256, 1) void k_gemm_bf16(
    const unsigned short* __restrict__ A, const unsigned short* __restrict__ B,
    float* __restrict__ C, int M, int N, int K) {
  __shared__ unsigned short As[2][BM * LDA_S];   // 2 x 10240 B
  __shared__ unsigned short Bs[2][BN * LDB_S];   // 2 x  5120 B, column-major Bs[b][n][k]

  int bm = blockIdx.y * BM;
  int bn = blockIdx.x * BN;
  int tid = threadIdx.x;
  int wave = tid >> 5;
  int wm = (wave & 3) * 32;    // 4 waves along M
  int wn = (wave >> 2) * 32;   // 2 waves along N

  f32x8 acc[2][2];
#pragma unroll
  for (int i = 0; i < 2; ++i)
#pragma unroll
    for (int j = 0; j < 2; ++j) acc[i][j] = (f32x8){0,0,0,0,0,0,0,0};

  // fixed per-thread load/stage coordinates
  int arow = tid >> 2;               // 0..63 (chunk0), +64 (chunk1)
  int acol = (tid & 3) * 8;          // 0,8,16,24
  int bk   = tid >> 3;               // 0..31
  int bn0  = (tid & 7) * 8;          // 0..56
  const unsigned short* aptr0 = A + (size_t)(bm + arow) * K + acol;
  const unsigned short* aptr1 = A + (size_t)(bm + 64 + arow) * K + acol;
  const unsigned short* bptr  = B + (size_t)bk * N + bn + bn0;

  int nsteps = K / BK;
  uint4 pa0 = *(const uint4*)aptr0;
  uint4 pa1 = *(const uint4*)aptr1;
  uint4 pb  = *(const uint4*)bptr;

  for (int s = 0; s < nsteps; ++s) {
    int buf = s & 1;
    // ---- stage prefetched tile into LDS ----
    *(uint4*)&As[buf][arow * LDA_S + acol]        = pa0;
    *(uint4*)&As[buf][(64 + arow) * LDA_S + acol] = pa1;
    {
      union { uint4 v4; unsigned short e[8]; } u; u.v4 = pb;
#pragma unroll
      for (int i = 0; i < 8; ++i) Bs[buf][(bn0 + i) * LDB_S + bk] = u.e[i];
    }
    __syncthreads();

    // ---- issue next tile's global loads (consumed next iteration) ----
    if (s + 1 < nsteps) {
      int k1 = (s + 1) * BK;
      pa0 = *(const uint4*)(aptr0 + k1);
      pa1 = *(const uint4*)(aptr1 + k1);
      pb  = *(const uint4*)(bptr + (size_t)k1 * N);
      if (s + 2 < nsteps) {                       // L2 prefetch at distance 2
        int k2 = (s + 2) * BK;
        __builtin_prefetch(aptr0 + k2, 0, 3);
        __builtin_prefetch(aptr1 + k2, 0, 3);
        __builtin_prefetch(bptr + (size_t)k2 * N, 0, 3);
      }
    }

    // ---- compute ----
    FragBF bfrag[2];
#pragma unroll
    for (int ni = 0; ni < 2; ++ni)
      bfrag[ni] = load_frag_b(&Bs[buf][(wn + ni * 16) * LDB_S], LDB_S);
#pragma unroll
    for (int mi = 0; mi < 2; ++mi) {
      FragBF a = load_frag_a(&As[buf][(wm + mi * 16) * LDA_S], LDA_S);
#pragma unroll
      for (int ni = 0; ni < 2; ++ni)
        acc[mi][ni] = wmma_bf16(a, bfrag[ni], acc[mi][ni]);
    }
    __syncthreads();
  }

  int lane = tid & 31;
  int n  = lane & 15;
  int mb = (lane < 16) ? 0 : 8;
#pragma unroll
  for (int mi = 0; mi < 2; ++mi)
#pragma unroll
    for (int ni = 0; ni < 2; ++ni) {
      float* c = C + (size_t)(bm + wm + mi * 16) * N + bn + wn + ni * 16;
#pragma unroll
      for (int r = 0; r < 8; ++r) c[(size_t)(mb + r) * N + n] = acc[mi][ni][r];
    }
}

// ---------- rms_norm: f32 in (ldx stride), bf16 out ----------
__global__ __launch_bounds__(256) void k_rmsnorm(
    const float* __restrict__ X, int ldx, const float* __restrict__ w,
    unsigned short* __restrict__ Y, int ldy, int N) {
  int row = blockIdx.x;
  const float* x = X + (size_t)row * ldx;
  float s = 0.f;
  for (int i = threadIdx.x; i < N; i += 256) { float v = x[i]; s += v * v; }
  __shared__ float red[256];
  red[threadIdx.x] = s; __syncthreads();
  for (int off = 128; off > 0; off >>= 1) {
    if (threadIdx.x < off) red[threadIdx.x] += red[threadIdx.x + off];
    __syncthreads();
  }
  float inv = rsqrtf(red[0] / (float)N + 1e-6f);
  unsigned short* y = Y + (size_t)row * ldy;
  for (int i = threadIdx.x; i < N; i += 256) y[i] = f2bf(x[i] * inv * w[i]);
}

// ---------- rope ----------
// permute_interleave + rotate_half:  out[j]=x[2j]c-x[2j+1]s ; out[32+j]=x[2j+1]c+x[2j]s
#define LN_BASE 17.0581027f   // ln(25600000)

__global__ void k_rope_q(const float* __restrict__ q, const int* __restrict__ pos,
                         unsigned short* __restrict__ qb) {
  int srow = blockIdx.x;
  const float* row = q + (size_t)srow * 6144;
  unsigned short* out = qb + (size_t)srow * 6144;
  float p = (float)pos[srow];
  for (int idx = threadIdx.x; idx < 6144; idx += blockDim.x) {
    int d = idx % 192;
    float v;
    if (d < 128) {
      v = row[idx];
    } else {
      int base = idx - d + 128;
      int j = (d - 128) & 31;
      float invf = __expf(-LN_BASE * (float)(2 * j) * (1.0f / 64.0f));
      float ang = p * invf;
      float c = __cosf(ang), sn = __sinf(ang);
      float x0 = row[base + 2 * j], x1 = row[base + 2 * j + 1];
      v = (d < 160) ? (x0 * c - x1 * sn) : (x1 * c + x0 * sn);
    }
    out[idx] = f2bf(v);
  }
}

__global__ void k_rope_kpe(const float* __restrict__ latent, const int* __restrict__ pos,
                           unsigned short* __restrict__ kpe) {
  int srow = blockIdx.x;
  const float* x = latent + (size_t)srow * 576 + 512;
  float p = (float)pos[srow];
  int j = threadIdx.x;
  if (j < 32) {
    float invf = __expf(-LN_BASE * (float)(2 * j) * (1.0f / 64.0f));
    float ang = p * invf;
    float c = __cosf(ang), sn = __sinf(ang);
    float x0 = x[2 * j], x1 = x[2 * j + 1];
    kpe[(size_t)srow * 64 + j]      = f2bf(x0 * c - x1 * sn);
    kpe[(size_t)srow * 64 + 32 + j] = f2bf(x1 * c + x0 * sn);
  }
}

// ---------- flash attention: per (q-tile of 64, head) ----------
#define LDQ 200
#define LDK 200
#define LDV 40
#define LDP 40

__global__ __launch_bounds__(128, 1) void k_attention(
    const unsigned short* __restrict__ qb,   // [2048][6144]  head h at col h*192
    const unsigned short* __restrict__ kvb,  // [2048][8192]  head h: nope@h*256, v@h*256+128
    const unsigned short* __restrict__ kpe,  // [2048][64]
    unsigned short* __restrict__ attn)       // [2048][4096]  head h at col h*128
{
  __shared__ unsigned short Qs[64 * LDQ];
  __shared__ unsigned short Ks[32 * LDK];
  __shared__ unsigned short Vt[128 * LDV];       // Vt[feat][key]
  __shared__ unsigned short Ps[4 * 16 * LDP];

  int qt = blockIdx.x, h = blockIdx.y;
  int q0 = qt * 64;
  int tid = threadIdx.x, wave = tid >> 5, lane = tid & 31;

  // stage Q tile (64 x 192)
  for (int idx = tid; idx < 64 * 24; idx += 128) {
    int r = idx / 24, cg = (idx % 24) * 8;
    uint4 v = *(const uint4*)(qb + (size_t)(q0 + r) * 6144 + h * 192 + cg);
    *(uint4*)&Qs[r * LDQ + cg] = v;
  }

  f32x8 o[8];
#pragma unroll
  for (int i = 0; i < 8; ++i) o[i] = (f32x8){0,0,0,0,0,0,0,0};
  float mrow[8], lrow[8];
#pragma unroll
  for (int r = 0; r < 8; ++r) { mrow[r] = -3.0e38f; lrow[r] = 0.f; }

  int myq = q0 + wave * 16 + ((lane < 16) ? 0 : 8);
  const float scale = 0.07216878364f;   // 192^-0.5
  int nkt = (qt + 1) * 2;

  for (int kt = 0; kt < nkt; ++kt) {
    int k0 = kt * 32;
    __syncthreads();
    // K tile 32 x 192: nope from kvb, rope feats from kpe
    for (int idx = tid; idx < 32 * 24; idx += 128) {
      int r = idx / 24, cg = (idx % 24) * 8;
      uint4 v;
      if (cg < 128) v = *(const uint4*)(kvb + (size_t)(k0 + r) * 8192 + h * 256 + cg);
      else          v = *(const uint4*)(kpe + (size_t)(k0 + r) * 64 + (cg - 128));
      *(uint4*)&Ks[r * LDK + cg] = v;
    }
    // V tile transposed: Vt[feat][key]
    for (int idx = tid; idx < 32 * 16; idx += 128) {
      int r = idx / 16, cg = (idx % 16) * 8;
      uint4 v = *(const uint4*)(kvb + (size_t)(k0 + r) * 8192 + h * 256 + 128 + cg);
      union { uint4 v4; unsigned short s[8]; } u; u.v4 = v;
#pragma unroll
      for (int i = 0; i < 8; ++i) Vt[(cg + i) * LDV + r] = u.s[i];
    }
    __syncthreads();

    // scores: 16x32 per wave = Q(16x192) @ K^T
    f32x8 sc[2];
    sc[0] = (f32x8){0,0,0,0,0,0,0,0};
    sc[1] = (f32x8){0,0,0,0,0,0,0,0};
#pragma unroll
    for (int ks = 0; ks < 6; ++ks) {
      FragBF a = load_frag_a(&Qs[(wave * 16) * LDQ + ks * 32], LDQ);
#pragma unroll
      for (int nt = 0; nt < 2; ++nt) {
        FragBF b = load_frag_b(&Ks[(nt * 16) * LDK + ks * 32], LDK);
        sc[nt] = wmma_bf16(a, b, sc[nt]);
      }
    }

    // scale + causal mask + online softmax (rows live on 16 lanes of one half)
    float mnew[8];
#pragma unroll
    for (int r = 0; r < 8; ++r) {
      int qrow = myq + r;
      float mx = mrow[r];
#pragma unroll
      for (int nt = 0; nt < 2; ++nt) {
        int kcol = k0 + nt * 16 + (lane & 15);
        float v = sc[nt][r] * scale + ((kcol > qrow) ? -1.0e9f : 0.0f);
        sc[nt][r] = v;
        mx = fmaxf(mx, v);
      }
      for (int msk = 1; msk < 16; msk <<= 1) mx = fmaxf(mx, __shfl_xor(mx, msk, 32));
      mnew[r] = mx;
    }
#pragma unroll
    for (int r = 0; r < 8; ++r) {
      float alpha = __expf(mrow[r] - mnew[r]);
      float ls = 0.f;
#pragma unroll
      for (int nt = 0; nt < 2; ++nt) {
        float pv = __expf(sc[nt][r] - mnew[r]);
        sc[nt][r] = pv;
        ls += pv;
      }
      for (int msk = 1; msk < 16; msk <<= 1) ls += __shfl_xor(ls, msk, 32);
      lrow[r] = lrow[r] * alpha + ls;
      mrow[r] = mnew[r];
#pragma unroll
      for (int i = 0; i < 8; ++i) o[i][r] *= alpha;
    }

    // re-layout P (C layout -> A layout) through private LDS patch
    unsigned short* myP = &Ps[wave * 16 * LDP];
#pragma unroll
    for (int nt = 0; nt < 2; ++nt) {
      int col = nt * 16 + (lane & 15);
#pragma unroll
      for (int r = 0; r < 8; ++r) {
        int row = (lane < 16) ? r : (8 + r);
        myP[row * LDP + col] = f2bf(sc[nt][r]);
      }
    }
    FragBF pa = load_frag_a(myP, LDP);          // same-wave DS ops are in order
#pragma unroll
    for (int ft = 0; ft < 8; ++ft) {
      FragBF b = load_frag_b(&Vt[(ft * 16) * LDV], LDV);
      o[ft] = wmma_bf16(pa, b, o[ft]);
    }
  }

  // epilogue: normalize, store bf16
#pragma unroll
  for (int ft = 0; ft < 8; ++ft)
#pragma unroll
    for (int r = 0; r < 8; ++r) {
      int qrow = myq + r;
      float val = o[ft][r] / lrow[r];
      attn[(size_t)qrow * 4096 + h * 128 + ft * 16 + (lane & 15)] = f2bf(val);
    }
}

// ---------- launch ----------
extern "C" void kernel_launch(void* const* d_in, const int* in_sizes, int n_in,
                              void* d_out, int out_size, void* d_ws, size_t ws_size,
                              hipStream_t stream) {
  (void)in_sizes; (void)n_in; (void)out_size; (void)ws_size;
  const float* hidden  = (const float*)d_in[0];
  const int*   pos     = (const int*)d_in[2];
  const float* Wq_a    = (const float*)d_in[3];
  const float* q_a_ln  = (const float*)d_in[4];
  const float* Wq_b    = (const float*)d_in[5];
  const float* Wkv_a   = (const float*)d_in[6];
  const float* kv_a_ln = (const float*)d_in[7];
  const float* Wkv_b   = (const float*)d_in[8];
  const float* Wo      = (const float*)d_in[9];
  float* out = (float*)d_out;

  char* ws = (char*)d_ws;
  size_t off = 0;
  auto alloc = [&](size_t bytes) -> void* {
    void* p = ws + off;
    off = (off + bytes + 255) & ~(size_t)255;
    return p;
  };
  unsigned short* h_bf    = (unsigned short*)alloc((size_t)2048 * 4096 * 2);
  unsigned short* wqa_bf  = (unsigned short*)alloc((size_t)4096 * 1536 * 2);
  unsigned short* wqb_bf  = (unsigned short*)alloc((size_t)1536 * 6144 * 2);
  unsigned short* wkva_bf = (unsigned short*)alloc((size_t)4096 * 576 * 2);
  unsigned short* wkvb_bf = (unsigned short*)alloc((size_t)512 * 8192 * 2);
  unsigned short* wo_bf   = (unsigned short*)alloc((size_t)4096 * 4096 * 2);
  unsigned short* q_a_n   = (unsigned short*)alloc((size_t)2048 * 1536 * 2);
  unsigned short* kv_a_n  = (unsigned short*)alloc((size_t)2048 * 512 * 2);
  unsigned short* kpe_bf  = (unsigned short*)alloc((size_t)2048 * 64 * 2);
  unsigned short* q_bf    = (unsigned short*)alloc((size_t)2048 * 6144 * 2);
  unsigned short* kv_bf   = (unsigned short*)alloc((size_t)2048 * 8192 * 2);
  unsigned short* attn_bf = (unsigned short*)alloc((size_t)2048 * 4096 * 2);
  // f32 arena, regions aliased over disjoint lifetimes
  char* arena = (char*)alloc((size_t)2048 * 8192 * 4);
  float* q_a_f    = (float*)arena;                              // steps 3-5
  float* q_f      = (float*)arena;                              // steps 8-9 (q_a dead)
  float* latent_f = (float*)(arena + (size_t)2048 * 6144 * 4);  // steps 4-7
  float* kv_f     = (float*)arena;                              // steps 10-11

  auto cvt = [&](const float* src, unsigned short* dst, size_t n) {
    size_t n4 = n / 4;
    k_f32_to_bf16_v4<<<dim3((unsigned)((n4 + 255) / 256)), 256, 0, stream>>>(src, dst, (int)n4);
  };
  cvt(hidden, h_bf,   (size_t)2048 * 4096);
  cvt(Wq_a,   wqa_bf, (size_t)4096 * 1536);
  cvt(Wq_b,   wqb_bf, (size_t)1536 * 6144);
  cvt(Wkv_a,  wkva_bf,(size_t)4096 * 576);
  cvt(Wkv_b,  wkvb_bf,(size_t)512 * 8192);
  cvt(Wo,     wo_bf,  (size_t)4096 * 4096);

  k_gemm_bf16<<<dim3(1536 / BN, 2048 / BM), 256, 0, stream>>>(h_bf, wqa_bf, q_a_f, 2048, 1536, 4096);
  k_gemm_bf16<<<dim3(576 / BN, 2048 / BM), 256, 0, stream>>>(h_bf, wkva_bf, latent_f, 2048, 576, 4096);
  k_rmsnorm<<<2048, 256, 0, stream>>>(q_a_f, 1536, q_a_ln, q_a_n, 1536, 1536);
  k_rmsnorm<<<2048, 256, 0, stream>>>(latent_f, 576, kv_a_ln, kv_a_n, 512, 512);
  k_rope_kpe<<<2048, 32, 0, stream>>>(latent_f, pos, kpe_bf);
  k_gemm_bf16<<<dim3(6144 / BN, 2048 / BM), 256, 0, stream>>>(q_a_n, wqb_bf, q_f, 2048, 6144, 1536);
  k_rope_q<<<2048, 256, 0, stream>>>(q_f, pos, q_bf);
  k_gemm_bf16<<<dim3(8192 / BN, 2048 / BM), 256, 0, stream>>>(kv_a_n, wkvb_bf, kv_f, 2048, 8192, 512);
  cvt(kv_f, kv_bf, (size_t)2048 * 8192);
  k_attention<<<dim3(32, 32), 128, 0, stream>>>(q_bf, kv_bf, kpe_bf, attn_bf);
  k_gemm_bf16<<<dim3(4096 / BN, 2048 / BM), 256, 0, stream>>>(attn_bf, wo_bf, out, 2048, 4096, 4096);
}
